// AESINDy_3753801416885
// MI455X (gfx1250) — matrix operand order repaired and checked by reading, not compiled, not run on hardware
//
#include <hip/hip_runtime.h>

typedef __attribute__((ext_vector_type(16))) _Float16 v16h;
typedef __attribute__((ext_vector_type(8)))  float    v8f;

#define ROWS_PER_WAVE 16
#define WAVES 8
#define ROWS_PER_BLOCK 128
#define LSIZE 220   // polynomial library size for 9 vars, order 3

// Compiler-only fence: LDS scratch is per-wave and DS ops are hardware
// in-order within a wave, so we only need to stop compiler reordering.
#define WAVE_LDS_FENCE() asm volatile("" ::: "memory")

// ---- WMMA fragment K-index maps (per CDNA5 ISA 7.12.2, 16-bit elements) ----
// A matrix 16x32: lane L holds row L%16; lanes<16: K in {0..7,16..23}, lanes>=16: {8..15,24..31}
__device__ __forceinline__ int kmapA(int i, int hi) {
    return ((i & 8) << 1) + (hi << 3) + (i & 7);
}
// B matrix 32x16: lane L holds col L%16; lanes<16: K in {0..7,16..23}, lanes>=16: {8..15,24..31}
__device__ __forceinline__ int kmapB(int i, int hi) {
    return ((i & 8) << 1) + (hi << 3) + (i & 7);
}

// Load a 16-half fragment from row-major f32 memory (converting), src pre-offset
// to (row-or-col)*ld + kstep*32 for this lane.
template <bool BMAT>
__device__ __forceinline__ v16h frag_from_f32(const float* __restrict__ src, int hi) {
    v16h f;
#pragma unroll
    for (int i = 0; i < 16; i += 2) {
        int k = BMAT ? kmapB(i, hi) : kmapA(i, hi);
        f[i]     = (_Float16)src[k];
        f[i + 1] = (_Float16)src[k + 1];
    }
    return f;
}

template <bool BMAT>
__device__ __forceinline__ v16h frag_from_lds(const _Float16* src, int hi) {
    v16h f;
#pragma unroll
    for (int i = 0; i < 16; i += 2) {
        int k = BMAT ? kmapB(i, hi) : kmapA(i, hi);
        f[i]     = src[k];
        f[i + 1] = src[k + 1];
    }
    return f;
}

__device__ __forceinline__ v8f wmma_f16(v16h a, v16h b, v8f c) {
    return __builtin_amdgcn_wmma_f32_16x16x32_f16(false, a, false, b, (short)0, c,
                                                  false, false);
}

__global__ __launch_bounds__(256, 1)
void aesindy_fused(const float* __restrict__ bin0,   // [B,256]
                   const float* __restrict__ Min,    // [B,1]
                   const float* __restrict__ w1, const float* __restrict__ b1,   // [128,256],[128]
                   const float* __restrict__ w2, const float* __restrict__ b2,   // [64,128],[64]
                   const float* __restrict__ w3, const float* __restrict__ b3,   // [32,64],[32]
                   const float* __restrict__ w4, const float* __restrict__ b4,   // [8,32],[8]
                   const float* __restrict__ sw,    // [9,220]
                   float* __restrict__ out)         // [B,8]
{
    __shared__ _Float16 sW1[128 * 256];  // 64 KB  (probing >64KB total static LDS)
    __shared__ _Float16 sW2[64 * 128];   // 16 KB
    __shared__ _Float16 sW3[32 * 64];    //  4 KB
    __shared__ _Float16 sW4[16 * 32];    //  1 KB (rows 8..15 zero-padded)
    __shared__ float sB1[128], sB2[64], sB3[32], sB4[16];
    __shared__ float sSW[8 * LSIZE];     // sindy rows 0..7, ~7 KB
    __shared__ _Float16 sAct[WAVES][16 * 128];  // per-wave activation scratch, 32 KB

    const int tid = threadIdx.x;

    // ---- stage weights (f32 -> f16) into LDS, once per block ----
    for (int i = tid; i < 128 * 256; i += 256) sW1[i] = (_Float16)w1[i];
    for (int i = tid; i < 64 * 128; i += 256)  sW2[i] = (_Float16)w2[i];
    for (int i = tid; i < 32 * 64; i += 256)   sW3[i] = (_Float16)w3[i];
    for (int i = tid; i < 16 * 32; i += 256)
        sW4[i] = (i < 8 * 32) ? (_Float16)w4[i] : (_Float16)0.0f;
    if (tid < 128) sB1[tid] = b1[tid];
    if (tid < 64)  sB2[tid] = b2[tid];
    if (tid < 32)  sB3[tid] = b3[tid];
    if (tid < 16)  sB4[tid] = (tid < 8) ? b4[tid] : 0.0f;
    for (int i = tid; i < 8 * LSIZE; i += 256) sSW[i] = sw[i];
    __syncthreads();   // only real cross-wave dependency in the kernel

    const int wave = tid >> 5;
    const int lane = tid & 31;
    const int l16  = lane & 15;
    const int hi   = lane >> 4;
    const int row0 = blockIdx.x * ROWS_PER_BLOCK + wave * ROWS_PER_WAVE;

    _Float16* act = &sAct[wave][0];

    // ================= layer 1: [16x256] @ [256x128] =================
    v16h a1[8];
#pragma unroll
    for (int ks = 0; ks < 8; ++ks)
        a1[ks] = frag_from_f32<false>(bin0 + (size_t)(row0 + l16) * 256 + ks * 32, hi);

#pragma unroll
    for (int nt = 0; nt < 8; ++nt) {
        v8f acc = {};
#pragma unroll
        for (int ks = 0; ks < 8; ++ks) {
            v16h bf = frag_from_lds<true>(sW1 + (nt * 16 + l16) * 256 + ks * 32, hi);
            acc = wmma_f16(a1[ks], bf, acc);
        }
        const int col = nt * 16 + l16;
        const float bias = sB1[col];
#pragma unroll
        for (int r = 0; r < 8; ++r) {
            float v = acc[r] + bias;
            v = v > 0.0f ? v : 0.0f;
            act[(r + hi * 8) * 128 + col] = (_Float16)v;
        }
    }
    WAVE_LDS_FENCE();

    // ================= layer 2: [16x128] @ [128x64] =================
    v16h a2[4];
#pragma unroll
    for (int ks = 0; ks < 4; ++ks)
        a2[ks] = frag_from_lds<false>(act + l16 * 128 + ks * 32, hi);
    WAVE_LDS_FENCE();
#pragma unroll
    for (int nt = 0; nt < 4; ++nt) {
        v8f acc = {};
#pragma unroll
        for (int ks = 0; ks < 4; ++ks) {
            v16h bf = frag_from_lds<true>(sW2 + (nt * 16 + l16) * 128 + ks * 32, hi);
            acc = wmma_f16(a2[ks], bf, acc);
        }
        const int col = nt * 16 + l16;
        const float bias = sB2[col];
#pragma unroll
        for (int r = 0; r < 8; ++r) {
            float v = acc[r] + bias;
            v = v > 0.0f ? v : 0.0f;
            act[(r + hi * 8) * 64 + col] = (_Float16)v;
        }
    }
    WAVE_LDS_FENCE();

    // ================= layer 3: [16x64] @ [64x32] =================
    v16h a3[2];
#pragma unroll
    for (int ks = 0; ks < 2; ++ks)
        a3[ks] = frag_from_lds<false>(act + l16 * 64 + ks * 32, hi);
    WAVE_LDS_FENCE();
#pragma unroll
    for (int nt = 0; nt < 2; ++nt) {
        v8f acc = {};
#pragma unroll
        for (int ks = 0; ks < 2; ++ks) {
            v16h bf = frag_from_lds<true>(sW3 + (nt * 16 + l16) * 64 + ks * 32, hi);
            acc = wmma_f16(a3[ks], bf, acc);
        }
        const int col = nt * 16 + l16;
        const float bias = sB3[col];
#pragma unroll
        for (int r = 0; r < 8; ++r) {
            float v = acc[r] + bias;
            v = v > 0.0f ? v : 0.0f;
            act[(r + hi * 8) * 32 + col] = (_Float16)v;
        }
    }
    WAVE_LDS_FENCE();

    // ================= layer 4: [16x32] @ [32x8->16] (linear) =================
    v16h a4 = frag_from_lds<false>(act + l16 * 32, hi);
    WAVE_LDS_FENCE();
    v16h b4f = frag_from_lds<true>(sW4 + l16 * 32, hi);
    v8f zacc = {};
    zacc = wmma_f16(a4, b4f, zacc);

    // write latent [16 rows][9] as f32 (z0..z7 + M) into per-wave scratch
    float* zbuf = (float*)act;   // 16*9 floats = 576 B << 4 KB slice
    if (l16 < 8) {
#pragma unroll
        for (int r = 0; r < 8; ++r)
            zbuf[(r + hi * 8) * 9 + l16] = zacc[r] + sB4[l16];
    } else if (l16 == 8) {
#pragma unroll
        for (int r = 0; r < 8; ++r) {
            int m = r + hi * 8;
            zbuf[m * 9 + 8] = Min[row0 + m];
        }
    }
    WAVE_LDS_FENCE();

    // ================= SINDy library + projection =================
    // 2 lanes per row; each lane computes 4 of the 8 retained outputs.
    {
        const int m  = lane >> 1;
        const int hh = lane & 1;
        float z[9];
#pragma unroll
        for (int i = 0; i < 9; ++i) z[i] = zbuf[m * 9 + i];

        const float* w0  = sSW + (hh * 4) * LSIZE;
        const float* wr1 = w0 + LSIZE;
        const float* wr2 = w0 + 2 * LSIZE;
        const float* wr3 = w0 + 3 * LSIZE;
        float o0, o1, o2, o3;
        int t = 0;
        // order 0: constant term (value 1)
        o0 = w0[t]; o1 = wr1[t]; o2 = wr2[t]; o3 = wr3[t]; ++t;
        // order 1
        for (int i = 0; i < 9; ++i) {
            float v = z[i];
            o0 += v * w0[t]; o1 += v * wr1[t]; o2 += v * wr2[t]; o3 += v * wr3[t];
            ++t;
        }
        // order 2: (i<=j) lexicographic
        for (int i = 0; i < 9; ++i) {
            float zi = z[i];
            for (int j = i; j < 9; ++j) {
                float v = zi * z[j];
                o0 += v * w0[t]; o1 += v * wr1[t]; o2 += v * wr2[t]; o3 += v * wr3[t];
                ++t;
            }
        }
        // order 3: (i<=j<=k) lexicographic
        for (int i = 0; i < 9; ++i) {
            float zi = z[i];
            for (int j = i; j < 9; ++j) {
                float zij = zi * z[j];
                for (int k = j; k < 9; ++k) {
                    float v = zij * z[k];
                    o0 += v * w0[t]; o1 += v * wr1[t]; o2 += v * wr2[t]; o3 += v * wr3[t];
                    ++t;
                }
            }
        }
        float4 res = make_float4(o0, o1, o2, o3);
        *(float4*)(out + (size_t)(row0 + m) * 8 + hh * 4) = res;
    }
}

extern "C" void kernel_launch(void* const* d_in, const int* in_sizes, int n_in,
                              void* d_out, int out_size, void* d_ws, size_t ws_size,
                              hipStream_t stream) {
    const float* bin0 = (const float*)d_in[0];
    const float* Min  = (const float*)d_in[1];
    const float* w1 = (const float*)d_in[2];
    const float* b1 = (const float*)d_in[3];
    const float* w2 = (const float*)d_in[4];
    const float* b2 = (const float*)d_in[5];
    const float* w3 = (const float*)d_in[6];
    const float* b3 = (const float*)d_in[7];
    const float* w4 = (const float*)d_in[8];
    const float* b4 = (const float*)d_in[9];
    const float* sw = (const float*)d_in[10];

    const int B = in_sizes[1];  // M has one element per row
    const int grid = B / ROWS_PER_BLOCK;
    aesindy_fused<<<grid, 256, 0, stream>>>(bin0, Min, w1, b1, w2, b2, w3, b3,
                                            w4, b4, sw, (float*)d_out);
    (void)d_ws; (void)ws_size; (void)n_in; (void)out_size;
}
